// GNN_l_h_45114336477554
// MI455X (gfx1250) — matrix operand order repaired and checked by reading, not compiled, not run on hardware
//
#include <hip/hip_runtime.h>
#include <hip/hip_bf16.h>

typedef _Float16 f16;
typedef __attribute__((ext_vector_type(16))) _Float16 v16h;
typedef __attribute__((ext_vector_type(8)))  float    v8f;

constexpr int F_DIMc = 13;
constexpr int MSG    = 32;
constexpr int HID    = 64;
constexpr int Bc     = 4;
constexpr int N_Lc   = 20000;
constexpr int N_Hc   = 100000;
constexpr int Ec     = 800000;
constexpr int IN_EDGE = 2 * F_DIMc + 8;   // 34
constexpr int IN_NODE = F_DIMc + MSG;     // 45
constexpr int KPAD    = 64;               // padded K for both layers

#define WMMA_F16(a, b, c) \
  __builtin_amdgcn_wmma_f32_16x16x32_f16(false, (a), false, (b), (short)0, (c), false, false)

// Fast transcendentals: keep everything on v_exp/v_rcp/v_tanh TRANS32 ops,
// never the precise-division (v_div_scale/v_div_fmas/v_div_fixup) sequence.
__device__ __forceinline__ float fast_tanh(float x) {
#if __has_builtin(__builtin_amdgcn_tanhf)
  return __builtin_amdgcn_tanhf(x);            // V_TANH_F32 (gfx1250 TRANS op)
#else
  const float e = __expf(-2.0f * x);           // v_mul + v_exp_f32
  const float t = __builtin_amdgcn_rcpf(1.0f + e);
  return __builtin_fmaf(-2.0f * e, t, 1.0f);   // tanh = 1 - 2e/(1+e)
#endif
}

__device__ __forceinline__ float fast_sigmoid(float x) {
  return __builtin_amdgcn_rcpf(1.0f + __expf(-x));
}

// A fragment (16x32 f16, M x K) from an LDS 16x64 row-major tile.
// ISA 7.12.2: lane l holds row m=l&15; half h <-> K = kstep*32 + 8*(l>>4) + (h&7) + 16*(h>>3)
__device__ __forceinline__ v16h afrag(const f16* __restrict__ base, int lane, int kstep) {
  const int m = lane & 15;
  const f16* row = base + m * KPAD + kstep * 32 + ((lane >> 4) << 3);
  v16h a;
#pragma unroll
  for (int h = 0; h < 16; ++h) a[h] = row[(h & 7) + ((h >> 3) << 4)];
  return a;
}

// B fragment (32x16 f16, K x N) from LDS weight matrix W[K][ldw] (row-major, zero-padded).
// lane l holds column n = ncol0 + (l&15); half h <-> K = kstep*32 + 8*(l>>4) + (h&7) + 16*(h>>3)
__device__ __forceinline__ v16h bfrag(const f16* __restrict__ W, int ldw, int lane,
                                      int kstep, int ncol0) {
  const int n = ncol0 + (lane & 15);
  const f16* col = W + (kstep * 32 + ((lane >> 4) << 3)) * ldw + n;
  v16h b;
#pragma unroll
  for (int h = 0; h < 16; ++h) b[h] = col[((h & 7) + ((h >> 3) << 4)) * ldw];
  return b;
}

// ---------------------------------------------------------------------------
// Kernel 0: zero the aggregation workspace (harness poisons d_ws)
// ---------------------------------------------------------------------------
__global__ void zero_f32_kernel(float* __restrict__ p, size_t n4) {
  size_t i = (size_t)blockIdx.x * blockDim.x + threadIdx.x;
  size_t stride = (size_t)gridDim.x * blockDim.x;
  float4* p4 = (float4*)p;
  float4 z = make_float4(0.f, 0.f, 0.f, 0.f);
  for (; i < n4; i += stride) p4[i] = z;
}

// ---------------------------------------------------------------------------
// Kernel 1: edge MLPs + weighted scatter-add.  One wave handles 16 edges/tile.
// ---------------------------------------------------------------------------
__global__ __launch_bounds__(256) void edge_kernel(
    const float* __restrict__ z_l, const float* __restrict__ z_h,
    const int* __restrict__ src, const int* __restrict__ tgt,
    const float* __restrict__ We1, const float* __restrict__ be1,
    const float* __restrict__ We2, const float* __restrict__ be2,
    const float* __restrict__ Ww1, const float* __restrict__ bw1,
    const float* __restrict__ Ww2, const float* __restrict__ bw2,
    float* __restrict__ agg) {
  __shared__ f16 sWe1[KPAD * HID];     // 34x64 padded to 64x64
  __shared__ f16 sWw1[KPAD * HID];
  __shared__ f16 sWe2[KPAD * MSG];     // 64x32
  __shared__ float sbe1[HID], sbw1[HID], sbe2[MSG], sWw2[HID];
  __shared__ f16 sFeat[8][16 * KPAD];  // per-wave edge-feature tile
  __shared__ f16 sH[8][16 * KPAD];     // per-wave tanh(hidden) tile
  __shared__ int sTgt[8][16];

  const int tid = threadIdx.x;
  // Stage zero-padded f16 weights in LDS (once per block).
  for (int i = tid; i < KPAD * HID; i += 256) {
    int r = i >> 6, c = i & 63;
    sWe1[i] = (r < IN_EDGE) ? (f16)We1[r * HID + c] : (f16)0.f;
    sWw1[i] = (r < IN_EDGE) ? (f16)Ww1[r * HID + c] : (f16)0.f;
  }
  for (int i = tid; i < KPAD * MSG; i += 256) {
    int r = i >> 5, c = i & 31;
    sWe2[i] = (f16)We2[r * MSG + c];   // K = 64 exactly
  }
  if (tid < HID) { sbe1[tid] = be1[tid]; sbw1[tid] = bw1[tid]; sWw2[tid] = Ww2[tid]; }
  if (tid < MSG) sbe2[tid] = be2[tid];
  __syncthreads();

  const int lane  = tid & 31;
  const int wslot = tid >> 5;
  const int nlo   = lane & 15;
  const int mhi   = (lane >> 4) << 3;
  const int wave   = (blockIdx.x * blockDim.x + tid) >> 5;
  const int nwaves = (gridDim.x * blockDim.x) >> 5;
  const int ntiles = (Bc * Ec) / 16;

  // Weight fragments live in registers across the tile loop.
  v16h bWe1[4][2], bWw1[4][2], bWe2[2][2];
#pragma unroll
  for (int t = 0; t < 4; ++t)
#pragma unroll
    for (int s = 0; s < 2; ++s) {
      bWe1[t][s] = bfrag(sWe1, HID, lane, s, 16 * t);
      bWw1[t][s] = bfrag(sWw1, HID, lane, s, 16 * t);
    }
#pragma unroll
  for (int t = 0; t < 2; ++t)
#pragma unroll
    for (int s = 0; s < 2; ++s) bWe2[t][s] = bfrag(sWe2, MSG, lane, s, 16 * t);

  const float biasw2 = bw2[0];
  f16* feat = sFeat[wslot];
  f16* hmat = sH[wslot];
  int* tgtl = sTgt[wslot];

  for (int tile = wave; tile < ntiles; tile += nwaves) {
    const int b = (tile * 16) / Ec;  // tiles never straddle batches (E % 16 == 0)

    // --- gather + geometric features -> LDS (lanes 0..15, one edge each) ---
    if (lane < 16) {
      const int gidx = tile * 16 + lane;
      const int si = src[gidx], ti = tgt[gidx];
      tgtl[lane] = ti;
      const float* zs = z_l + ((size_t)b * N_Lc + si) * F_DIMc;
      const float* zt = z_h + ((size_t)b * N_Hc + ti) * F_DIMc;
      float a[13], c[13];
#pragma unroll
      for (int k = 0; k < 13; ++k) { a[k] = zs[k]; c[k] = zt[k]; }
      const float dx = a[0] - c[0], dy = a[1] - c[1], dz = a[2] - c[2];
      const float dist = dx * dx + dy * dy + dz * dz;
      const float cx = a[4] * c[5] - a[5] * c[4];
      const float cy = a[5] * c[3] - a[3] * c[5];
      const float cz = a[3] * c[4] - a[4] * c[3];
      const float an = sqrtf(cx * cx + cy * cy + cz * cz);
      f16* row = feat + lane * KPAD;
#pragma unroll
      for (int k = 0; k < 13; ++k) { row[k] = (f16)a[k]; row[13 + k] = (f16)c[k]; }
      row[26] = (f16)dx; row[27] = (f16)dy; row[28] = (f16)dz; row[29] = (f16)dist;
      row[30] = (f16)cx; row[31] = (f16)cy; row[32] = (f16)cz; row[33] = (f16)an;
#pragma unroll
      for (int k = IN_EDGE; k < KPAD; ++k) row[k] = (f16)0.f;
    }
    asm volatile("s_wait_dscnt 0" ::: "memory");

    const v16h a0 = afrag(feat, lane, 0);
    const v16h a1 = afrag(feat, lane, 1);

    // --- gate MLP: hidden -> tanh -> dot(Ww2) -> sigmoid (reg r <-> edge r+mhi) ---
    float wsum[8];
#pragma unroll
    for (int r = 0; r < 8; ++r) wsum[r] = 0.f;
#pragma unroll
    for (int t = 0; t < 4; ++t) {
      v8f acc;
      const float bini = sbw1[16 * t + nlo];
#pragma unroll
      for (int r = 0; r < 8; ++r) acc[r] = bini;
      acc = WMMA_F16(a0, bWw1[t][0], acc);
      acc = WMMA_F16(a1, bWw1[t][1], acc);
      const float w2 = sWw2[16 * t + nlo];
#pragma unroll
      for (int r = 0; r < 8; ++r) wsum[r] += fast_tanh(acc[r]) * w2;
    }
#pragma unroll
    for (int r = 0; r < 8; ++r) {
#pragma unroll
      for (int msk = 1; msk < 16; msk <<= 1) wsum[r] += __shfl_xor(wsum[r], msk, 32);
      wsum[r] = fast_sigmoid(wsum[r] + biasw2);  // w for edge m = r + mhi
    }

    // --- message MLP layer 1: hidden -> tanh -> LDS (for A relayout) ---
#pragma unroll
    for (int t = 0; t < 4; ++t) {
      v8f acc;
      const float bini = sbe1[16 * t + nlo];
#pragma unroll
      for (int r = 0; r < 8; ++r) acc[r] = bini;
      acc = WMMA_F16(a0, bWe1[t][0], acc);
      acc = WMMA_F16(a1, bWe1[t][1], acc);
#pragma unroll
      for (int r = 0; r < 8; ++r)
        hmat[(r + mhi) * KPAD + 16 * t + nlo] = (f16)fast_tanh(acc[r]);
    }
    asm volatile("s_wait_dscnt 0" ::: "memory");

    const v16h h0 = afrag(hmat, lane, 0);
    const v16h h1 = afrag(hmat, lane, 1);
    float* aggb = agg + (size_t)b * N_Hc * MSG;

    // --- message MLP layer 2 + weighted scatter-add ---
#pragma unroll
    for (int t = 0; t < 2; ++t) {
      v8f acc;
      const float bini = sbe2[16 * t + nlo];
#pragma unroll
      for (int r = 0; r < 8; ++r) acc[r] = bini;
      acc = WMMA_F16(h0, bWe2[t][0], acc);
      acc = WMMA_F16(h1, bWe2[t][1], acc);
      const int n = 16 * t + nlo;
#pragma unroll
      for (int r = 0; r < 8; ++r) {
        const int m = r + mhi;
        const float v = wsum[r] * acc[r];
        __hip_atomic_fetch_add(&aggb[(size_t)tgtl[m] * MSG + n], v,
                               __ATOMIC_RELAXED, __HIP_MEMORY_SCOPE_AGENT);
      }
    }
  }
}

// ---------------------------------------------------------------------------
// Kernel 2: node MLP over concat(z_h, agg).  One wave handles 16 nodes/tile.
// ---------------------------------------------------------------------------
__global__ __launch_bounds__(256) void node_kernel(
    const float* __restrict__ z_h, const float* __restrict__ agg,
    const float* __restrict__ Wn1, const float* __restrict__ bn1,
    const float* __restrict__ Wn2, const float* __restrict__ bn2,
    float* __restrict__ out) {
  __shared__ f16 sWn1[KPAD * HID];   // 45x64 padded to 64x64
  __shared__ f16 sWn2[KPAD * 16];    // 64x13 padded to 64x16
  __shared__ float sbn1[HID], sbn2[16];
  __shared__ f16 sFeat[8][16 * KPAD];
  __shared__ f16 sH[8][16 * KPAD];

  const int tid = threadIdx.x;
  for (int i = tid; i < KPAD * HID; i += 256) {
    int r = i >> 6, c = i & 63;
    sWn1[i] = (r < IN_NODE) ? (f16)Wn1[r * HID + c] : (f16)0.f;
  }
  for (int i = tid; i < KPAD * 16; i += 256) {
    int r = i >> 4, c = i & 15;
    sWn2[i] = (c < F_DIMc) ? (f16)Wn2[r * F_DIMc + c] : (f16)0.f;
  }
  if (tid < HID) sbn1[tid] = bn1[tid];
  if (tid < 16)  sbn2[tid] = (tid < F_DIMc) ? bn2[tid] : 0.f;
  __syncthreads();

  const int lane  = tid & 31;
  const int wslot = tid >> 5;
  const int nlo   = lane & 15;
  const int mhi   = (lane >> 4) << 3;
  const int wave   = (blockIdx.x * blockDim.x + tid) >> 5;
  const int nwaves = (gridDim.x * blockDim.x) >> 5;
  const int ntiles = (Bc * N_Hc) / 16;

  v16h bWn1[4][2], bWn2[2];
#pragma unroll
  for (int t = 0; t < 4; ++t)
#pragma unroll
    for (int s = 0; s < 2; ++s) bWn1[t][s] = bfrag(sWn1, HID, lane, s, 16 * t);
#pragma unroll
  for (int s = 0; s < 2; ++s) bWn2[s] = bfrag(sWn2, 16, lane, s, 0);

  f16* feat = sFeat[wslot];
  f16* hmat = sH[wslot];

  for (int tile = wave; tile < ntiles; tile += nwaves) {
    if (lane < 16) {
      const size_t gn = (size_t)tile * 16 + lane;   // flat (b, node)
      const float* zh = z_h + gn * F_DIMc;
      const float* ag = agg + gn * MSG;
      f16* row = feat + lane * KPAD;
#pragma unroll
      for (int k = 0; k < F_DIMc; ++k) row[k] = (f16)zh[k];
#pragma unroll
      for (int k = 0; k < MSG; ++k) row[F_DIMc + k] = (f16)ag[k];
#pragma unroll
      for (int k = IN_NODE; k < KPAD; ++k) row[k] = (f16)0.f;
    }
    asm volatile("s_wait_dscnt 0" ::: "memory");

    const v16h a0 = afrag(feat, lane, 0);
    const v16h a1 = afrag(feat, lane, 1);

#pragma unroll
    for (int t = 0; t < 4; ++t) {
      v8f acc;
      const float bini = sbn1[16 * t + nlo];
#pragma unroll
      for (int r = 0; r < 8; ++r) acc[r] = bini;
      acc = WMMA_F16(a0, bWn1[t][0], acc);
      acc = WMMA_F16(a1, bWn1[t][1], acc);
#pragma unroll
      for (int r = 0; r < 8; ++r)
        hmat[(r + mhi) * KPAD + 16 * t + nlo] = (f16)fast_tanh(acc[r]);
    }
    asm volatile("s_wait_dscnt 0" ::: "memory");

    const v16h h0 = afrag(hmat, lane, 0);
    const v16h h1 = afrag(hmat, lane, 1);

    v8f acc;
    const float bini = sbn2[nlo];
#pragma unroll
    for (int r = 0; r < 8; ++r) acc[r] = bini;
    acc = WMMA_F16(h0, bWn2[0], acc);
    acc = WMMA_F16(h1, bWn2[1], acc);

    if (nlo < F_DIMc) {
#pragma unroll
      for (int r = 0; r < 8; ++r) {
        const size_t gn = (size_t)tile * 16 + r + mhi;
        out[gn * F_DIMc + nlo] = acc[r];
      }
    }
  }
}

// ---------------------------------------------------------------------------
extern "C" void kernel_launch(void* const* d_in, const int* in_sizes, int n_in,
                              void* d_out, int out_size, void* d_ws, size_t ws_size,
                              hipStream_t stream) {
  const float* z_l = (const float*)d_in[0];
  const float* z_h = (const float*)d_in[1];
  const int*   src = (const int*)d_in[2];
  const int*   tgt = (const int*)d_in[3];
  const float* We1 = (const float*)d_in[4];
  const float* be1 = (const float*)d_in[5];
  const float* We2 = (const float*)d_in[6];
  const float* be2 = (const float*)d_in[7];
  const float* Ww1 = (const float*)d_in[8];
  const float* bw1 = (const float*)d_in[9];
  const float* Ww2 = (const float*)d_in[10];
  const float* bw2 = (const float*)d_in[11];
  const float* Wn1 = (const float*)d_in[12];
  const float* bn1 = (const float*)d_in[13];
  const float* Wn2 = (const float*)d_in[14];
  const float* bn2 = (const float*)d_in[15];

  float* agg = (float*)d_ws;                       // (B, N_H, MSG) f32
  float* out = (float*)d_out;                      // (B, N_H, F_DIM) f32
  const size_t aggN = (size_t)Bc * N_Hc * MSG;     // 12.8M floats

  zero_f32_kernel<<<dim3(2048), dim3(256), 0, stream>>>(agg, aggN / 4);
  edge_kernel<<<dim3(2048), dim3(256), 0, stream>>>(
      z_l, z_h, src, tgt, We1, be1, We2, be2, Ww1, bw1, Ww2, bw2, agg);
  node_kernel<<<dim3(1024), dim3(256), 0, stream>>>(
      z_h, agg, Wn1, bn1, Wn2, bn2, out);
}
